// EuclideanCodebook_24189255811228
// MI455X (gfx1250) — matrix operand order repaired and checked by reading, not compiled, not run on hardware
//
#include <hip/hip_runtime.h>

// ---------------------------------------------------------------------------
// VQ Euclidean codebook search for MI455X (gfx1250, wave32, WMMA)
//   x:     [1, 8, 4096, 512] f32  -> flat M=32768 x D=512
//   embed: [1, 8192, 512]    f32  -> C=8192 x D=512
// Outputs (concatenated in d_out, all f32):
//   quantize [M*512] | embed_ind [M] | dist [M*C]
// ---------------------------------------------------------------------------

typedef __attribute__((ext_vector_type(16))) __bf16 v16bf;
typedef __attribute__((ext_vector_type(8)))  float  v8f;

static constexpr int M_TOTAL = 32768;   // 8 * 4096
static constexpr int C_TOTAL = 8192;
static constexpr int D_DIM   = 512;
static constexpr int TILE    = 128;     // 128x128 dist tile per workgroup
static constexpr int KSTEP   = 32;      // K per WMMA step (bf16 16x16x32)
static constexpr int BF_STRIDE = 40;    // bf16 row stride = 20 dwords ->
                                        // 16 rows hit 16 distinct 4-dword
                                        // bank groups: conflict-free b128
static constexpr size_t KEYS_OFF = (size_t)(M_TOTAL + C_TOTAL) * sizeof(float);

// ---------------------------------------------------------------------------
// Kernel 1: row norms for x and embed; re-init argmin keys (every launch).
// One wave per row; 512/32 = 16 elems per lane, shfl_xor reduction.
// ---------------------------------------------------------------------------
__global__ void __launch_bounds__(256)
vq_norms_kernel(const float* __restrict__ x,
                const float* __restrict__ embed,
                float* __restrict__ x2,
                float* __restrict__ y2,
                unsigned long long* __restrict__ keys)
{
    const int row  = blockIdx.x * 8 + (threadIdx.x >> 5);
    const int lane = threadIdx.x & 31;

    const float* src;
    float* dst;
    if (row < M_TOTAL) {
        src = x + (size_t)row * D_DIM;
        dst = x2 + row;
    } else {
        const int r = row - M_TOTAL;
        src = embed + (size_t)r * D_DIM;
        dst = y2 + r;
    }

    float s = 0.0f;
#pragma unroll
    for (int i = 0; i < D_DIM / 32; ++i) {
        const float v = src[lane + i * 32];
        s = fmaf(v, v, s);
    }
#pragma unroll
    for (int off = 16; off; off >>= 1)
        s += __shfl_xor(s, off, 32);
    if (lane == 0) *dst = s;

    // reset per-row argmin keys (atomicMin targets) — must happen each call
    const int gid = blockIdx.x * (int)blockDim.x + (int)threadIdx.x;
    if (gid < M_TOTAL) keys[gid] = 0xFFFFFFFFFFFFFFFFull;
}

// ---------------------------------------------------------------------------
// Kernel 2: dist tile GEMM. f32 tiles are split into bf16 hi/lo ONCE during
// LDS staging; the WMMA inner loop is pure ds_load_b128 + v_wmma.
// Workgroup = 256 threads = 8 waves; each wave owns 16 rows x 128 cols.
// ---------------------------------------------------------------------------
__global__ void __launch_bounds__(256)
vq_dist_kernel(const float* __restrict__ x,
               const float* __restrict__ embed,
               const float* __restrict__ x2,
               const float* __restrict__ y2,
               float* __restrict__ dist_out,
               unsigned long long* __restrict__ keys)
{
    __shared__ __bf16 Ah[TILE][BF_STRIDE];
    __shared__ __bf16 Al[TILE][BF_STRIDE];
    __shared__ __bf16 Bh[TILE][BF_STRIDE];
    __shared__ __bf16 Bl[TILE][BF_STRIDE];

    const int tid  = threadIdx.x;
    const int lane = tid & 31;
    const int wave = tid >> 5;
    const int l15  = lane & 15;     // N index within 16x16 subtile / A row
    const int lhi  = lane >> 4;     // lane half (selects K group / M half)

    const int nBlock = blockIdx.x * TILE;
    const int mBlock = blockIdx.y * TILE;

    union Acc   { v8f v;    float  f[8];  };
    union Frag  { v16bf v;  uint4  u[2];  };
    union Stage { float4 v4[4]; float f[16]; };
    union Pack8 { __bf16 h[8]; uint4 u; };

    Acc acc[8];
#pragma unroll
    for (int nt = 0; nt < 8; ++nt)
#pragma unroll
        for (int i = 0; i < 8; ++i) acc[nt].f[i] = 0.0f;

    // cooperative staging: thread t covers row t/2, 16 floats (4x float4)
    const int ldRow = tid >> 1;
    const int ldCol = (tid & 1) * 16;
    const float* aPtr = x     + (size_t)(mBlock + ldRow) * D_DIM + ldCol;
    const float* bPtr = embed + (size_t)(nBlock + ldRow) * D_DIM + ldCol;

    Stage aS, bS;
#pragma unroll
    for (int i = 0; i < 4; ++i) {           // prologue: k-step 0
        aS.v4[i] = ((const float4*)aPtr)[i];
        bS.v4[i] = ((const float4*)bPtr)[i];
    }

    for (int ks = 0; ks < D_DIM; ks += KSTEP) {
        // ---- hi/lo split happens here ONCE per element per workgroup ----
        Pack8 ah[2], al[2], bh[2], bl[2];
#pragma unroll
        for (int i = 0; i < 16; ++i) {
            const float va = aS.f[i];
            const float vb = bS.f[i];
            const __bf16 ha = (__bf16)va;
            const __bf16 hb = (__bf16)vb;
            ah[i >> 3].h[i & 7] = ha;
            bh[i >> 3].h[i & 7] = hb;
            al[i >> 3].h[i & 7] = (__bf16)(va - (float)ha);
            bl[i >> 3].h[i & 7] = (__bf16)(vb - (float)hb);
        }
        *(uint4*)&Ah[ldRow][ldCol]     = ah[0].u;
        *(uint4*)&Ah[ldRow][ldCol + 8] = ah[1].u;
        *(uint4*)&Al[ldRow][ldCol]     = al[0].u;
        *(uint4*)&Al[ldRow][ldCol + 8] = al[1].u;
        *(uint4*)&Bh[ldRow][ldCol]     = bh[0].u;
        *(uint4*)&Bh[ldRow][ldCol + 8] = bh[1].u;
        *(uint4*)&Bl[ldRow][ldCol]     = bl[0].u;
        *(uint4*)&Bl[ldRow][ldCol + 8] = bl[1].u;
        __syncthreads();

        // software pipeline: issue next k-step's global loads now; they
        // complete while the WMMAs below execute.
        if (ks + KSTEP < D_DIM) {
#pragma unroll
            for (int i = 0; i < 4; ++i) {
                aS.v4[i] = ((const float4*)(aPtr + ks + KSTEP))[i];
                bS.v4[i] = ((const float4*)(bPtr + ks + KSTEP))[i];
            }
        }

        // A fragment (16x32 bf16), contiguous K order (matches B):
        //   lanes 0-15:  h[0..15]=K0..15   (M = lane)
        //   lanes 16-31: h[0..15]=K16..31  (M = lane-16)
        const int am  = wave * 16 + l15;
        const int akb = lhi * 16;
        Frag aHi, aLo;
        aHi.u[0] = *(const uint4*)&Ah[am][akb];
        aHi.u[1] = *(const uint4*)&Ah[am][akb + 8];
        aLo.u[0] = *(const uint4*)&Al[am][akb];
        aLo.u[1] = *(const uint4*)&Al[am][akb + 8];

#pragma unroll
        for (int nt = 0; nt < 8; ++nt) {
            // B fragment (32x16 bf16): lanes 0-15 hold K0..15 (N = lane),
            // lanes 16-31 hold K16..31; contiguous per lane.
            const int bn  = nt * 16 + l15;
            const int bkb = lhi * 16;
            Frag bHi, bLo;
            bHi.u[0] = *(const uint4*)&Bh[bn][bkb];
            bHi.u[1] = *(const uint4*)&Bh[bn][bkb + 8];
            bLo.u[0] = *(const uint4*)&Bl[bn][bkb];
            bLo.u[1] = *(const uint4*)&Bl[bn][bkb + 8];
            // xy ~= hi*hi + hi*lo + lo*hi  (drop lo*lo; ~2^-16 rel err)
            acc[nt].v = __builtin_amdgcn_wmma_f32_16x16x32_bf16(
                false, aHi.v, false, bHi.v, (short)0, acc[nt].v, false, false);
            acc[nt].v = __builtin_amdgcn_wmma_f32_16x16x32_bf16(
                false, aHi.v, false, bLo.v, (short)0, acc[nt].v, false, false);
            acc[nt].v = __builtin_amdgcn_wmma_f32_16x16x32_bf16(
                false, aLo.v, false, bHi.v, (short)0, acc[nt].v, false, false);
        }
        __syncthreads();
    }

    // ---- epilogue: d2 = x2 + y2 - 2*xy ; dist = -sqrt(d2) ; fused argmin ----
    // D layout: lane (l15) = N, vgpr v = M - 8*lhi within the 16x16 subtile.
    float x2r[8], y2r[8];
#pragma unroll
    for (int v = 0; v < 8; ++v)
        x2r[v] = x2[mBlock + wave * 16 + lhi * 8 + v];
#pragma unroll
    for (int nt = 0; nt < 8; ++nt)
        y2r[nt] = y2[nBlock + nt * 16 + l15];

    unsigned long long key[8];
#pragma unroll
    for (int v = 0; v < 8; ++v) key[v] = 0xFFFFFFFFFFFFFFFFull;

#pragma unroll
    for (int nt = 0; nt < 8; ++nt) {
        const int nG = nBlock + nt * 16 + l15;
#pragma unroll
        for (int v = 0; v < 8; ++v) {
            const int mG = mBlock + wave * 16 + lhi * 8 + v;
            float d2 = fmaf(-2.0f, acc[nt].f[v], x2r[v] + y2r[nt]);
            d2 = fmaxf(d2, 0.0f);
            // streamed 1 GB output: non-temporal, keep L2 for x/embed tiles
            __builtin_nontemporal_store(-sqrtf(d2),
                                        dist_out + (size_t)mG * C_TOTAL + nG);
            // non-negative f32 bit pattern is order-preserving;
            // ties resolve to the lowest column == argmax-first semantics
            const unsigned long long k =
                ((unsigned long long)__float_as_uint(d2) << 32) | (unsigned)nG;
            key[v] = key[v] < k ? key[v] : k;
        }
    }

    // min-reduce across the 16 lanes sharing each row (xor masks stay in-half)
#pragma unroll
    for (int v = 0; v < 8; ++v) {
#pragma unroll
        for (int off = 8; off; off >>= 1) {
            const unsigned long long o = __shfl_xor(key[v], off, 32);
            key[v] = key[v] < o ? key[v] : o;
        }
    }
    if (l15 == 0) {
#pragma unroll
        for (int v = 0; v < 8; ++v)
            atomicMin(&keys[mBlock + wave * 16 + lhi * 8 + v], key[v]);
    }
}

// ---------------------------------------------------------------------------
// Kernel 3: decode argmin index, emit embed_ind (as float) + gather codebook.
// One wave per row; 512 floats = 4x float4 per lane.
// ---------------------------------------------------------------------------
__global__ void __launch_bounds__(256)
vq_gather_kernel(const float* __restrict__ embed,
                 const unsigned long long* __restrict__ keys,
                 float* __restrict__ quant_out,
                 float* __restrict__ ind_out)
{
    const int row  = blockIdx.x * 8 + (threadIdx.x >> 5);
    const int lane = threadIdx.x & 31;

    const unsigned idx = (unsigned)(keys[row] & 0xFFFFFFFFu);
    if (lane == 0) ind_out[row] = (float)idx;

    const float4* src = (const float4*)(embed + (size_t)idx * D_DIM);
    float4*       dst = (float4*)(quant_out + (size_t)row * D_DIM);
#pragma unroll
    for (int j = 0; j < 4; ++j)
        dst[lane + 32 * j] = src[lane + 32 * j];
}

// ---------------------------------------------------------------------------
extern "C" void kernel_launch(void* const* d_in, const int* in_sizes, int n_in,
                              void* d_out, int out_size, void* d_ws, size_t ws_size,
                              hipStream_t stream)
{
    const float* x     = (const float*)d_in[0];   // [1,8,4096,512]
    const float* embed = (const float*)d_in[1];   // [1,8192,512]
    float* out = (float*)d_out;

    // workspace: x2[M] | y2[C] | keys[M] (u64)   (~416 KB)
    float* x2 = (float*)d_ws;
    float* y2 = x2 + M_TOTAL;
    unsigned long long* keys =
        (unsigned long long*)((char*)d_ws + KEYS_OFF);

    // d_out: quantize [M*512] | embed_ind [M] | dist [M*C]
    float* quant_out = out;
    float* ind_out   = out + (size_t)M_TOTAL * D_DIM;
    float* dist_out  = ind_out + M_TOTAL;

    vq_norms_kernel<<<(M_TOTAL + C_TOTAL) / 8, 256, 0, stream>>>(
        x, embed, x2, y2, keys);

    dim3 grid(C_TOTAL / TILE, M_TOTAL / TILE);   // 64 x 256 workgroups
    vq_dist_kernel<<<grid, 256, 0, stream>>>(
        x, embed, x2, y2, dist_out, keys);

    vq_gather_kernel<<<M_TOTAL / 8, 256, 0, stream>>>(
        embed, keys, quant_out, ind_out);
}
